// BioHAN_37022618091686
// MI455X (gfx1250) — matrix operand order repaired and checked by reading, not compiled, not run on hardware
//
#include <hip/hip_runtime.h>
#include <hip/hip_bf16.h>

// ---------------------------------------------------------------------------
// BioHAN forward for MI455X (gfx1250, wave32, WMMA + async-to-LDS).
// Shapes: B=2048, G=10000, H=256, 2H=512, C=16, S=32, M=100, L=3
//
// Exact simplifications:
//  * seq-len==1 attention => softmax over size-1 axis == 1 => o == v.
//  * sparsity_loss == 0.01/G == 1e-6 exactly.
//  * marker_scores is a pure broadcast (1.31 GB of stores = the roofline term).
// ---------------------------------------------------------------------------

typedef __attribute__((ext_vector_type(16))) __bf16 v16bf;
typedef __attribute__((ext_vector_type(8)))  __bf16 v8bf;
typedef __attribute__((ext_vector_type(8)))  float  v8f;

#define A_PITCH 36   // f32 per As row (144B: b128-aligned, conflict-free reads)
#define B_PITCH 40   // bf16 per Bs row (80B: b128-aligned, low conflict)

// ---------------------------------------------------------------------------
// C[M,N] = A[M,K] @ B[K,N] + bias[N], bf16 WMMA, f32 accumulate.
// Block: 256 threads (8 waves, 2x4) -> 64x64 output tile.
// Wave: 32x16 output (2 WMMAs/chunk, B fragment reused across both).
// A tile (64x32 f32) staged via GLOBAL_LOAD_ASYNC_TO_LDS_B128 (ASYNCcnt);
// B tile (32x64) loaded as float4, stored transposed bf16 -> ds_load_b128 frags.
// Double-buffered, one barrier per 32-wide K chunk. M%64==0, N%64==0 required.
// ---------------------------------------------------------------------------
__global__ __launch_bounds__(256) void gemm_bf16_wmma(
    const float* __restrict__ A, const float* __restrict__ Bm,
    const float* __restrict__ bias, float* __restrict__ C,
    int M, int N, int K, int lda, int ldb, int ldc)
{
    __shared__ float  Asf[2][64 * A_PITCH];   // 18432 B
    __shared__ __bf16 Bs [2][64 * B_PITCH];   // 10240 B

    const int t    = threadIdx.x;
    const int wid  = t >> 5;
    const int lane = t & 31;
    const int r    = lane & 15;
    const int half = lane >> 4;

    const int m0 = blockIdx.y * 64;
    const int n0 = blockIdx.x * 64;

    // A loader: 64 rows x 32 k, 8 consecutive f32/thread (two async b128)
    const int la_m = t >> 2;          // 0..63
    const int la_k = (t & 3) * 8;     // 0,8,16,24
    // B loader: 32 k x 64 n, 8 consecutive n/thread (two float4 loads)
    const int lb_k = t >> 3;          // 0..31
    const int lb_n = (t & 7) * 8;     // 0,8,...,56

    const int nchunk = (K + 31) >> 5;

    auto load_tile = [&](int c, int buf) {
        const int kk = c * 32;
        const float* gsrc = A + (size_t)(m0 + la_m) * lda + kk + la_k;
        float* ldst = &Asf[buf][la_m * A_PITCH + la_k];
        const float* bp = Bm + (size_t)(kk + lb_k) * ldb + n0 + lb_n;
        __bf16* bdst = &Bs[buf][0];

        if (kk + 32 <= K) {
            // ---- fast path ----
            const unsigned d0 = (unsigned)(uintptr_t)ldst;   // LDS byte offset
            const unsigned d1 = d0 + 16;
            asm volatile("global_load_async_to_lds_b128 %0, %1, off"
                         :: "v"(d0), "v"(gsrc) : "memory");
            asm volatile("global_load_async_to_lds_b128 %0, %1, off"
                         :: "v"(d1), "v"(gsrc + 4) : "memory");

            const float4 q0 = *reinterpret_cast<const float4*>(bp);
            const float4 q1 = *reinterpret_cast<const float4*>(bp + 4);
            bdst[(lb_n + 0) * B_PITCH + lb_k] = (__bf16)q0.x;
            bdst[(lb_n + 1) * B_PITCH + lb_k] = (__bf16)q0.y;
            bdst[(lb_n + 2) * B_PITCH + lb_k] = (__bf16)q0.z;
            bdst[(lb_n + 3) * B_PITCH + lb_k] = (__bf16)q0.w;
            bdst[(lb_n + 4) * B_PITCH + lb_k] = (__bf16)q1.x;
            bdst[(lb_n + 5) * B_PITCH + lb_k] = (__bf16)q1.y;
            bdst[(lb_n + 6) * B_PITCH + lb_k] = (__bf16)q1.z;
            bdst[(lb_n + 7) * B_PITCH + lb_k] = (__bf16)q1.w;
        } else {
            // ---- K tail: predicated, zero fill ----
            #pragma unroll
            for (int j = 0; j < 8; ++j)
                ldst[j] = (kk + la_k + j < K) ? gsrc[j] : 0.0f;
            const bool kok = (kk + lb_k) < K;
            #pragma unroll
            for (int j = 0; j < 8; ++j)
                bdst[(lb_n + j) * B_PITCH + lb_k] =
                    kok ? (__bf16)bp[j] : (__bf16)0.0f;
        }
    };

    load_tile(0, 0);
    asm volatile("s_wait_asynccnt 0" ::: "memory");
    __syncthreads();

    v8f acc0 = {}, acc1 = {};
    const int miBase = (wid >> 2) * 32 + r;   // As rows: miBase, miBase+16
    const int ni     = (wid & 3) * 16 + r;    // Bs row (output column)

    for (int c = 0; c < nchunk; ++c) {
        const int buf = c & 1;
        if (c + 1 < nchunk) load_tile(c + 1, buf ^ 1);

        // B fragment: col = ni, k = half*8 + j and 16 + half*8 + j
        const v8bf* pb = reinterpret_cast<const v8bf*>(&Bs[buf][ni * B_PITCH + half * 8]);
        v8bf b0 = pb[0], b1 = pb[2];
        v16bf bf;
        #pragma unroll
        for (int j = 0; j < 8; ++j) { bf[j] = b0[j]; bf[j + 8] = b1[j]; }

        // Two A fragments (rows miBase, miBase+16), reuse B for both WMMAs
        #pragma unroll
        for (int a = 0; a < 2; ++a) {
            const int mi = miBase + a * 16;
            const float4* pa0 = reinterpret_cast<const float4*>(&Asf[buf][mi * A_PITCH + half * 8]);
            const float4* pa1 = reinterpret_cast<const float4*>(&Asf[buf][mi * A_PITCH + 16 + half * 8]);
            float4 qa0 = pa0[0], qa1 = pa0[1];
            float4 qa2 = pa1[0], qa3 = pa1[1];
            v16bf af;
            af[0]  = (__bf16)qa0.x; af[1]  = (__bf16)qa0.y; af[2]  = (__bf16)qa0.z; af[3]  = (__bf16)qa0.w;
            af[4]  = (__bf16)qa1.x; af[5]  = (__bf16)qa1.y; af[6]  = (__bf16)qa1.z; af[7]  = (__bf16)qa1.w;
            af[8]  = (__bf16)qa2.x; af[9]  = (__bf16)qa2.y; af[10] = (__bf16)qa2.z; af[11] = (__bf16)qa2.w;
            af[12] = (__bf16)qa3.x; af[13] = (__bf16)qa3.y; af[14] = (__bf16)qa3.z; af[15] = (__bf16)qa3.w;
            if (a == 0)
                acc0 = __builtin_amdgcn_wmma_f32_16x16x32_bf16(
                           false, af, false, bf, (short)0, acc0, false, false);
            else
                acc1 = __builtin_amdgcn_wmma_f32_16x16x32_bf16(
                           false, af, false, bf, (short)0, acc1, false, false);
        }

        asm volatile("s_wait_asynccnt 0" ::: "memory");
        __syncthreads();
    }

    const int gm = m0 + (wid >> 2) * 32;
    const int gn = n0 + (wid & 3) * 16 + r;
    const float bv = bias ? bias[gn] : 0.0f;
    #pragma unroll
    for (int v = 0; v < 8; ++v)
        C[(size_t)(gm + half * 8 + v) * ldc + gn] = acc0[v] + bv;
    #pragma unroll
    for (int v = 0; v < 8; ++v)
        C[(size_t)(gm + 16 + half * 8 + v) * ldc + gn] = acc1[v] + bv;
}

// ---------------------------------------------------------------------------
// LayerNorm over last dim (256 or 512), optional residual add and ReLU.
// ---------------------------------------------------------------------------
__global__ __launch_bounds__(256) void ln_kernel(
    const float* __restrict__ X, const float* __restrict__ R,
    const float* __restrict__ g, const float* __restrict__ b,
    float* __restrict__ out, int H, int do_relu)
{
    __shared__ float red[16];
    __shared__ float stats[2];
    const int row = blockIdx.x, t = threadIdx.x;
    const int ept = H >> 8;

    float xv[2];
    float s = 0.0f, s2 = 0.0f;
    for (int e = 0; e < ept; ++e) {
        int idx = e * 256 + t;
        float v = X[(size_t)row * H + idx];
        if (R) v += R[(size_t)row * H + idx];
        xv[e] = v; s += v; s2 += v * v;
    }
    #pragma unroll
    for (int off = 16; off; off >>= 1) {
        s  += __shfl_xor(s,  off, 32);
        s2 += __shfl_xor(s2, off, 32);
    }
    if ((t & 31) == 0) { red[t >> 5] = s; red[8 + (t >> 5)] = s2; }
    __syncthreads();
    if (t == 0) {
        float S = 0.0f, S2 = 0.0f;
        for (int i = 0; i < 8; ++i) { S += red[i]; S2 += red[8 + i]; }
        float m = S / (float)H;
        stats[0] = m;
        stats[1] = rsqrtf(S2 / (float)H - m * m + 1e-5f);
    }
    __syncthreads();
    const float m = stats[0], inv = stats[1];
    for (int e = 0; e < ept; ++e) {
        int idx = e * 256 + t;
        float y = (xv[e] - m) * inv * g[idx] + b[idx];
        if (do_relu) y = fmaxf(y, 0.0f);
        out[(size_t)row * H + idx] = y;
    }
}

// ---------------------------------------------------------------------------
__global__ __launch_bounds__(256) void proto_norm_kernel(
    const float* __restrict__ protos, float* __restrict__ pn)
{
    const int t = threadIdx.x, lane = t & 31, wid = t >> 5;
    for (int ss = 0; ss < 4; ++ss) {
        int s = wid * 4 + ss;
        float d = 0.0f;
        #pragma unroll
        for (int j = 0; j < 8; ++j) {
            float v = protos[s * 256 + lane + 32 * j];
            d += v * v;
        }
        #pragma unroll
        for (int off = 16; off; off >>= 1) d += __shfl_xor(d, off, 32);
        if (lane == 0) pn[s] = sqrtf(d);
    }
}

// ---------------------------------------------------------------------------
// Head: cosine sim -> softmax(T=0.5) -> cell_states -> 2-layer MLP logits.
// One block per sample.
// ---------------------------------------------------------------------------
__global__ __launch_bounds__(256) void head_kernel(
    const float* __restrict__ f, const float* __restrict__ protos,
    const float* __restrict__ pn,
    const float* __restrict__ w1, const float* __restrict__ b1,
    const float* __restrict__ w2, const float* __restrict__ b2,
    float* __restrict__ logits, float* __restrict__ cs_out)
{
    __shared__ float feats[256];
    __shared__ float cs[256];
    __shared__ float hidden[128];
    __shared__ float sw[32];
    __shared__ float red[8];
    __shared__ float fn_sh;

    const int b = blockIdx.x, t = threadIdx.x;
    const int lane = t & 31, wid = t >> 5;

    const float fv = f[(size_t)b * 256 + t];
    feats[t] = fv;
    __syncthreads();

    float p = fv * fv;
    #pragma unroll
    for (int off = 16; off; off >>= 1) p += __shfl_xor(p, off, 32);
    if (lane == 0) red[wid] = p;
    __syncthreads();
    if (t == 0) {
        float s = 0.0f;
        for (int i = 0; i < 8; ++i) s += red[i];
        fn_sh = sqrtf(s);
    }
    __syncthreads();

    for (int ss = 0; ss < 4; ++ss) {
        int s = wid * 4 + ss;
        float d = 0.0f;
        #pragma unroll
        for (int j = 0; j < 8; ++j)
            d += feats[lane + 32 * j] * protos[s * 256 + lane + 32 * j];
        #pragma unroll
        for (int off = 16; off; off >>= 1) d += __shfl_xor(d, off, 32);
        if (lane == 0) sw[s] = d;
    }
    __syncthreads();

    if (t < 32) {
        float sim = sw[t] / fmaxf(fn_sh * pn[t], 1e-8f);
        float xx = sim * 2.0f;  // / STATE_TEMP(0.5)
        float mx = xx;
        #pragma unroll
        for (int off = 16; off; off >>= 1) mx = fmaxf(mx, __shfl_xor(mx, off, 32));
        float e = __expf(xx - mx);
        float sum = e;
        #pragma unroll
        for (int off = 16; off; off >>= 1) sum += __shfl_xor(sum, off, 32);
        sw[t] = e / sum;
    }
    __syncthreads();

    float c = 0.0f;
    #pragma unroll 8
    for (int s = 0; s < 32; ++s) c += sw[s] * protos[s * 256 + t];
    cs[t] = c;
    cs_out[(size_t)b * 256 + t] = c;
    __syncthreads();

    if (t < 128) {
        float h = b1[t];
        for (int j = 0; j < 256; ++j) h += cs[j] * w1[j * 128 + t];
        hidden[t] = fmaxf(h, 0.0f);
    }
    __syncthreads();

    if (t < 16) {
        float l = b2[t];
        for (int j = 0; j < 128; ++j) l += hidden[j] * w2[j * 16 + t];
        logits[(size_t)b * 16 + t] = l;
    }
}

// ---------------------------------------------------------------------------
__global__ void sparsity_kernel(float* __restrict__ out)
{
    if (threadIdx.x == 0 && blockIdx.x == 0) out[0] = 1e-6f;  // == 0.01/G exactly
}

// ---------------------------------------------------------------------------
// marker_scores[b,c,g] = marker_weights[c,g]: 1.31 GB broadcast, nontemporal.
// ---------------------------------------------------------------------------
__global__ __launch_bounds__(256) void broadcast_markers(
    const float* __restrict__ w, float* __restrict__ out)
{
    const int idx = blockIdx.x * 256 + threadIdx.x;
    if (idx >= 16 * 10000) return;
    const float val = w[idx];
    float* o = out + idx;
    for (int b = 0; b < 2048; ++b)
        __builtin_nontemporal_store(val, o + (size_t)b * 160000);
}

// ---------------------------------------------------------------------------
__global__ __launch_bounds__(256) void marker_importance_kernel(
    const float* __restrict__ w, float* __restrict__ out)
{
    __shared__ float red[8];
    __shared__ float denom_sh;
    const int c = blockIdx.x, t = threadIdx.x;
    const int lane = t & 31, wid = t >> 5;
    float p = 0.0f;
    for (int g = t; g < 10000; g += 256) p += fabsf(w[c * 10000 + g]);
    #pragma unroll
    for (int off = 16; off; off >>= 1) p += __shfl_xor(p, off, 32);
    if (lane == 0) red[wid] = p;
    __syncthreads();
    if (t == 0) {
        float s = 0.0f;
        for (int i = 0; i < 8; ++i) s += red[i];
        denom_sh = fmaxf(s, 1e-12f);
    }
    __syncthreads();
    const float inv = 1.0f / denom_sh;
    for (int g = t; g < 10000; g += 256)
        out[c * 10000 + g] = w[c * 10000 + g] * inv;
}

// ---------------------------------------------------------------------------
extern "C" void kernel_launch(void* const* d_in, const int* in_sizes, int n_in,
                              void* d_out, int out_size, void* d_ws, size_t ws_size,
                              hipStream_t stream)
{
    const float* x        = (const float*)d_in[0];
    const float* enc_w1   = (const float*)d_in[1];
    const float* enc_b1   = (const float*)d_in[2];
    const float* enc_ln_g = (const float*)d_in[3];
    const float* enc_ln_b = (const float*)d_in[4];
    const float* enc_w2   = (const float*)d_in[5];
    const float* enc_b2   = (const float*)d_in[6];
    // d_in[7] mod_assign, d_in[8] mod_inter: only feed sparsity_loss (==1e-6)
    const float* qkv_w    = (const float*)d_in[9];
    const float* qkv_b    = (const float*)d_in[10];
    const float* out_w    = (const float*)d_in[11];
    const float* out_b    = (const float*)d_in[12];
    const float* ln_g     = (const float*)d_in[13];
    const float* ln_b     = (const float*)d_in[14];
    const float* protos   = (const float*)d_in[15];
    const float* clf_w1   = (const float*)d_in[16];
    const float* clf_b1   = (const float*)d_in[17];
    const float* clf_w2   = (const float*)d_in[18];
    const float* clf_b2   = (const float*)d_in[19];
    const float* markers  = (const float*)d_in[20];

    float* out      = (float*)d_out;
    float* logits_o = out;                 // 2048*16
    float* cs_o     = out + 32768;         // 2048*256
    float* sp_o     = out + 557056;        // 1
    float* ms_o     = out + 557057;        // 2048*16*10000
    float* mi_o     = out + 328237057;     // 160000

    float* w      = (float*)d_ws;
    float* h_buf  = w;                     // 2048*512
    float* f_buf  = w + 1048576;           // 2048*256
    float* v_buf  = w + 1572864;           // 2048*256
    float* o_buf  = w + 2097152;           // 2048*256
    float* pn_buf = w + 2621440;           // 32

    const dim3 blk(256);

    gemm_bf16_wmma<<<dim3(8, 32), blk, 0, stream>>>(
        x, enc_w1, enc_b1, h_buf, 2048, 512, 10000, 10000, 512, 512);
    ln_kernel<<<2048, blk, 0, stream>>>(
        h_buf, nullptr, enc_ln_g, enc_ln_b, h_buf, 512, 1);
    gemm_bf16_wmma<<<dim3(4, 32), blk, 0, stream>>>(
        h_buf, enc_w2, enc_b2, f_buf, 2048, 256, 512, 512, 256, 256);

    for (int i = 0; i < 3; ++i) {
        gemm_bf16_wmma<<<dim3(4, 32), blk, 0, stream>>>(
            f_buf, qkv_w + (size_t)i * 256 * 768 + 512, qkv_b + i * 768 + 512,
            v_buf, 2048, 256, 256, 256, 768, 256);
        gemm_bf16_wmma<<<dim3(4, 32), blk, 0, stream>>>(
            v_buf, out_w + (size_t)i * 256 * 256, out_b + i * 256,
            o_buf, 2048, 256, 256, 256, 256, 256);
        ln_kernel<<<2048, blk, 0, stream>>>(
            o_buf, f_buf, ln_g + i * 256, ln_b + i * 256, f_buf, 256, 0);
    }

    proto_norm_kernel<<<1, blk, 0, stream>>>(protos, pn_buf);
    head_kernel<<<2048, blk, 0, stream>>>(
        f_buf, protos, pn_buf, clf_w1, clf_b1, clf_w2, clf_b2, logits_o, cs_o);

    sparsity_kernel<<<1, 1, 0, stream>>>(sp_o);
    broadcast_markers<<<625, blk, 0, stream>>>(markers, ms_o);
    marker_importance_kernel<<<16, blk, 0, stream>>>(markers, mi_o);
}
